// PromptGenerator_62654982914315
// MI455X (gfx1250) — matrix operand order repaired and checked by reading, not compiled, not run on hardware
//
#include <hip/hip_runtime.h>
#include <cstdint>
#include <cstddef>

// Problem dims (fixed by the reference)
#define B_ 4096
#define D_ 768
#define C_ 512
#define L_ 32
#define H_ 256
#define E_ (C_*L_)   // 16384

typedef __attribute__((ext_vector_type(16))) __bf16 v16bf;
typedef __attribute__((ext_vector_type(8)))  __bf16 v8bf;
typedef __attribute__((ext_vector_type(8)))  float  v8f;

#define HAS_ASYNC_LDS (__has_builtin(__builtin_amdgcn_global_load_async_to_lds_b128) && \
                       __has_builtin(__builtin_amdgcn_s_wait_asynccnt))

#if HAS_ASYNC_LDS
typedef int v4i_ __attribute__((vector_size(16)));
typedef __attribute__((address_space(1))) v4i_* async_gp_t;   // global int4*
typedef __attribute__((address_space(3))) v4i_* async_lp_t;   // LDS int4*
#endif

// ---------- helpers ----------

static __device__ __forceinline__ unsigned short f2bf(float f) {
  union { float f; uint32_t u; } x; x.f = f;
  uint32_t r = x.u + 0x7FFFu + ((x.u >> 16) & 1u);   // round-to-nearest-even
  return (unsigned short)(r >> 16);
}

// block-wide reduction over 256 threads (8 waves of 32). red must have >= 9 floats.
static __device__ __forceinline__ float blockReduce(float v, float* red, bool isMax) {
  #pragma unroll
  for (int off = 16; off; off >>= 1) {
    float o = __shfl_xor(v, off, 32);
    v = isMax ? fmaxf(v, o) : (v + o);
  }
  if ((threadIdx.x & 31) == 0) red[threadIdx.x >> 5] = v;
  __syncthreads();
  if (threadIdx.x == 0) {
    float s = red[0];
    for (int i = 1; i < 8; ++i) s = isMax ? fmaxf(s, red[i]) : (s + red[i]);
    red[8] = s;
  }
  __syncthreads();
  float r = red[8];
  __syncthreads();
  return r;
}

// ---------- kernel 0: ext_embs fp32 -> bf16 (row-major kept) ----------
__global__ void k_cvt_ext(const float* __restrict__ src,
                          unsigned short* __restrict__ dst, int n4) {
  int i = blockIdx.x * blockDim.x + threadIdx.x;
  if (i < n4) {
    float4 v = ((const float4*)src)[i];
    ushort4 o;
    o.x = f2bf(v.x); o.y = f2bf(v.y); o.z = f2bf(v.z); o.w = f2bf(v.w);
    ((ushort4*)dst)[i] = o;
  }
}

// ---------- kernel 1: q = l2norm(text+img); write fp32 and bf16 ----------
__global__ void k_qprep(const float* __restrict__ text, const float* __restrict__ img,
                        float* __restrict__ qf, unsigned short* __restrict__ qb) {
  __shared__ float red[9];
  const int b = blockIdx.x;
  const int t = threadIdx.x;
  const size_t base = (size_t)b * D_;
  float v0 = text[base + t]       + img[base + t];
  float v1 = text[base + t + 256] + img[base + t + 256];
  float v2 = text[base + t + 512] + img[base + t + 512];
  float ss = blockReduce(v0*v0 + v1*v1 + v2*v2, red, false);
  float inv = 1.0f / fmaxf(sqrtf(ss), 1e-12f);
  v0 *= inv; v1 *= inv; v2 *= inv;
  qf[base + t] = v0; qf[base + t + 256] = v1; qf[base + t + 512] = v2;
  qb[base + t] = f2bf(v0); qb[base + t + 256] = f2bf(v1); qb[base + t + 512] = f2bf(v2);
}

// ---------- kernel 2: sim GEMM (bf16 WMMA, f32 acc) fused with segment max ----------
// grid = (B/32, C/8), block = 256 (8 waves). Wave w -> concept c0+w (32 ext rows).
// Each wave: 2 M-tiles x 2 N-tiles (4 accumulators) so every B tile is used twice.
__global__ void __launch_bounds__(256)
k_simmax(const unsigned short* __restrict__ qb,
         const unsigned short* __restrict__ eb,
         float* __restrict__ prior) {
  __shared__ unsigned short qs[32 * D_];          // 48 KB: 32 q rows, bf16
  const int tid  = threadIdx.x;
  const int m0   = blockIdx.x * 32;
  const int c    = blockIdx.y * 8 + (tid >> 5);
  const int lane = tid & 31;
  const int n    = lane & 15;                     // column within 16-wide N tile
  const int hi   = lane >> 4;                     // lane half selects K chunk

  // stage the 32x768 q tile (contiguous 48 KB) into LDS
#if HAS_ASYNC_LDS
  {
    async_gp_t gsrc = (async_gp_t)(qb + (size_t)m0 * D_ + (size_t)tid * 8);
    async_lp_t ldst = (async_lp_t)(&qs[tid * 8]);
    // 48 KB total; 256 threads * 16 B = 4 KB per issue -> 12 async issues
    #pragma unroll
    for (int i = 0; i < 12; ++i)
      __builtin_amdgcn_global_load_async_to_lds_b128(gsrc + i * 256, ldst + i * 256, 0, 0);
    __builtin_amdgcn_s_wait_asynccnt(0);
  }
#else
  {
    const uint4* s = (const uint4*)(qb + (size_t)m0 * D_);
    uint4* d = (uint4*)qs;
    #pragma unroll
    for (int i = tid; i < 32 * D_ / 8; i += 256) d[i] = s[i];
  }
#endif
  __syncthreads();

  // A operands: row m = lane&15 (+16 for second M tile);
  // lanes<16 hold K {0..15}, lanes>=16 hold K {16..31}
  const __bf16* arow0 = ((const __bf16*)qs) + n * D_ + hi * 16;
  const __bf16* arow1 = arow0 + 16 * D_;
  // B operands: lane = column n -> ext row (c*32 + n); 16 contiguous K per lane,
  // lanes 16-31 take K+16. Second N tile = ext rows +16.
  const __bf16* b0p = ((const __bf16*)eb) + ((size_t)c * L_ + n) * D_ + hi * 16;
  const __bf16* b1p = b0p + (size_t)16 * D_;

  v8f acc00 = {}; v8f acc01 = {};                 // M tile 0 x N tiles {0,1}
  v8f acc10 = {}; v8f acc11 = {};                 // M tile 1 x N tiles {0,1}
  #pragma unroll 2
  for (int k = 0; k < D_; k += 32) {
    v8bf a0lo = *(const v8bf*)(arow0 + k);
    v8bf a0hi = *(const v8bf*)(arow0 + k + 8);
    v8bf a1lo = *(const v8bf*)(arow1 + k);
    v8bf a1hi = *(const v8bf*)(arow1 + k + 8);
    v16bf a0, a1;
    #pragma unroll
    for (int i = 0; i < 8; ++i) {
      a0[i] = a0lo[i]; a0[i + 8] = a0hi[i];
      a1[i] = a1lo[i]; a1[i + 8] = a1hi[i];
    }
    v16bf bm0 = *(const v16bf*)(b0p + k);
    v16bf bm1 = *(const v16bf*)(b1p + k);
    __builtin_prefetch(b0p + k + 128, 0, 3);
    __builtin_prefetch(b1p + k + 128, 0, 3);
    acc00 = __builtin_amdgcn_wmma_f32_16x16x32_bf16(false, a0, false, bm0,
                                                    (short)0, acc00, false, false);
    acc01 = __builtin_amdgcn_wmma_f32_16x16x32_bf16(false, a0, false, bm1,
                                                    (short)0, acc01, false, false);
    acc10 = __builtin_amdgcn_wmma_f32_16x16x32_bf16(false, a1, false, bm0,
                                                    (short)0, acc10, false, false);
    acc11 = __builtin_amdgcn_wmma_f32_16x16x32_bf16(false, a1, false, bm1,
                                                    (short)0, acc11, false, false);
  }

  // segment max over the concept's 32 columns:
  // elementwise max of the two N tiles, then max across lanes 0-15 / 16-31.
  float mxA[8], mxB[8];
  #pragma unroll
  for (int r = 0; r < 8; ++r) {
    mxA[r] = fmaxf(acc00[r], acc01[r]);
    mxB[r] = fmaxf(acc10[r], acc11[r]);
  }
  #pragma unroll
  for (int off = 1; off < 16; off <<= 1) {
    #pragma unroll
    for (int r = 0; r < 8; ++r) {
      mxA[r] = fmaxf(mxA[r], __shfl_xor(mxA[r], off, 16));
      mxB[r] = fmaxf(mxB[r], __shfl_xor(mxB[r], off, 16));
    }
  }
  // C/D layout: VGPR r, lanes 0-15 -> M=r ; lanes 16-31 -> M=r+8
  if (n == 0) {
    const int mrow = m0 + hi * 8;
    #pragma unroll
    for (int r = 0; r < 8; ++r) {
      prior[(size_t)(mrow + r) * C_ + c]      = mxA[r];
      prior[(size_t)(mrow + 16 + r) * C_ + c] = mxB[r];
    }
  }
}

// ---------- kernel 3: h = relu([prior,q] @ W1 + b1), 16-row blocks ----------
__global__ void __launch_bounds__(256)
k_mlp1(const float* __restrict__ prior, const float* __restrict__ qf,
       const float* __restrict__ W1, const float* __restrict__ b1,
       float* __restrict__ h) {
  __shared__ float xs[16 * 256];                  // 16 KB k-chunk of x
  const int tid = threadIdx.x;
  const int m0  = blockIdx.x * 16;
  float acc[16];
  #pragma unroll
  for (int r = 0; r < 16; ++r) acc[r] = 0.f;

  for (int chunk = 0; chunk < 5; ++chunk) {       // K = 1280 = 5 * 256
    const int kbase = chunk * 256;
    for (int i = tid; i < 16 * 256; i += 256) {
      int r = i >> 8, kk = i & 255, k = kbase + kk;
      float v = (k < C_) ? prior[(size_t)(m0 + r) * C_ + k]
                         : qf[(size_t)(m0 + r) * D_ + (k - C_)];
      xs[i] = v;
    }
    __syncthreads();
    for (int kk = 0; kk < 256; ++kk) {
      float w = W1[(size_t)(kbase + kk) * H_ + tid];
      #pragma unroll
      for (int r = 0; r < 16; ++r) acc[r] = fmaf(xs[r * 256 + kk], w, acc[r]);
    }
    __syncthreads();
  }
  #pragma unroll
  for (int r = 0; r < 16; ++r)
    h[(size_t)(m0 + r) * H_ + tid] = fmaxf(acc[r] + b1[tid], 0.f);
}

// ---------- kernel 4: logits = h@W2+b2; softmax(logits/T) -> weights ----------
__global__ void __launch_bounds__(256)
k_soft(const float* __restrict__ h, const float* __restrict__ W2,
       const float* __restrict__ b2, const float* __restrict__ tempp,
       float* __restrict__ wts) {
  __shared__ float hs[16 * H_];                   // 16 KB
  __shared__ float ls[16 * C_];                   // 32 KB (logits / T)
  __shared__ float red[9];
  const int tid = threadIdx.x;
  const int m0  = blockIdx.x * 16;
  const float invT = 1.0f / tempp[0];

  for (int i = tid; i < 16 * H_; i += 256)
    hs[i] = h[(size_t)m0 * H_ + i];
  __syncthreads();

  #pragma unroll
  for (int g = 0; g < 2; ++g) {                   // columns tid and tid+256
    const int cc = tid + g * 256;
    float acc[16];
    #pragma unroll
    for (int r = 0; r < 16; ++r) acc[r] = 0.f;
    for (int k = 0; k < H_; ++k) {
      float w = W2[(size_t)k * C_ + cc];
      #pragma unroll
      for (int r = 0; r < 16; ++r) acc[r] = fmaf(hs[r * H_ + k], w, acc[r]);
    }
    #pragma unroll
    for (int r = 0; r < 16; ++r) ls[r * C_ + cc] = (acc[r] + b2[cc]) * invT;
  }
  __syncthreads();

  for (int r = 0; r < 16; ++r) {
    float a0 = ls[r * C_ + tid], a1 = ls[r * C_ + tid + 256];
    float m  = blockReduce(fmaxf(a0, a1), red, true);
    float e0 = __expf(a0 - m), e1 = __expf(a1 - m);
    float S  = blockReduce(e0 + e1, red, false);
    float inv = 1.0f / S;
    wts[(size_t)(m0 + r) * C_ + tid]       = e0 * inv;
    wts[(size_t)(m0 + r) * C_ + tid + 256] = e1 * inv;
  }
}

// ---------- kernel 5: dyn = weights@concept_names; out = l2norm(text+a*dyn) ----------
__global__ void __launch_bounds__(256)
k_dyn(const float* __restrict__ wts, const float* __restrict__ concept,
      const float* __restrict__ text, const float* __restrict__ alphap,
      float* __restrict__ out) {
  __shared__ float wsld[16 * C_];                 // 32 KB weights tile
  __shared__ float red[9];
  const int tid = threadIdx.x;
  const int m0  = blockIdx.x * 16;
  const float alpha = alphap[0];

  for (int i = tid; i < 16 * C_; i += 256)
    wsld[i] = wts[(size_t)m0 * C_ + i];
  __syncthreads();

  float comb[3][16];
  float ssum[16];
  #pragma unroll
  for (int r = 0; r < 16; ++r) ssum[r] = 0.f;

  #pragma unroll
  for (int g = 0; g < 3; ++g) {                   // D = 768 = 3 * 256 columns
    const int d = tid + g * 256;
    float acc[16];
    #pragma unroll
    for (int r = 0; r < 16; ++r) acc[r] = 0.f;
    for (int cc = 0; cc < C_; ++cc) {
      float cn = concept[(size_t)cc * D_ + d];
      #pragma unroll
      for (int r = 0; r < 16; ++r) acc[r] = fmaf(wsld[r * C_ + cc], cn, acc[r]);
    }
    #pragma unroll
    for (int r = 0; r < 16; ++r) {
      float v = text[(size_t)(m0 + r) * D_ + d] + alpha * acc[r];
      comb[g][r] = v;
      ssum[r] = fmaf(v, v, ssum[r]);
    }
  }
  __syncthreads();

  float invn[16];
  for (int r = 0; r < 16; ++r) {
    float S = blockReduce(ssum[r], red, false);
    invn[r] = 1.0f / fmaxf(sqrtf(S), 1e-12f);
  }
  #pragma unroll
  for (int g = 0; g < 3; ++g)
    #pragma unroll
    for (int r = 0; r < 16; ++r)
      out[(size_t)(m0 + r) * D_ + tid + g * 256] = comb[g][r] * invn[r];
}

// ---------- launch ----------
extern "C" void kernel_launch(void* const* d_in, const int* in_sizes, int n_in,
                              void* d_out, int out_size, void* d_ws, size_t ws_size,
                              hipStream_t stream) {
  const float* text    = (const float*)d_in[0];
  const float* img     = (const float*)d_in[1];
  const float* ext     = (const float*)d_in[2];
  // d_in[3] = segment_ids (structure is a fixed repeat of L=32; not needed)
  const float* concept = (const float*)d_in[4];
  const float* W1      = (const float*)d_in[5];
  const float* b1      = (const float*)d_in[6];
  const float* W2      = (const float*)d_in[7];
  const float* b2      = (const float*)d_in[8];
  const float* alphap  = (const float*)d_in[9];
  const float* tempp   = (const float*)d_in[10];
  float* out = (float*)d_out;

  // workspace layout (256B-aligned offsets)
  char* ws = (char*)d_ws;
  size_t off = 0;
  auto alloc = [&](size_t bytes) { char* p = ws + off; off = (off + bytes + 255) & ~(size_t)255; return p; };
  float*          qf    = (float*)         alloc((size_t)B_ * D_ * 4);
  unsigned short* qb    = (unsigned short*)alloc((size_t)B_ * D_ * 2);
  unsigned short* eb    = (unsigned short*)alloc((size_t)E_ * D_ * 2);
  float*          prior = (float*)         alloc((size_t)B_ * C_ * 4);
  float*          hbuf  = (float*)         alloc((size_t)B_ * H_ * 4);
  float*          wts   = (float*)         alloc((size_t)B_ * C_ * 4);
  (void)ws_size; (void)in_sizes; (void)n_in; (void)out_size;

  // 0) ext fp32 -> bf16 (25 MB out, memory-bound, ~4 us)
  {
    int n4 = E_ * D_ / 4;
    k_cvt_ext<<<(n4 + 255) / 256, 256, 0, stream>>>(ext, eb, n4);
  }
  // 1) q = l2norm(text+img)
  k_qprep<<<B_, 256, 0, stream>>>(text, img, qf, qb);
  // 2) 103-GFLOP GEMM + segment max via bf16 WMMA (32 rows x 8 concepts per block)
  k_simmax<<<dim3(B_ / 32, C_ / 8), 256, 0, stream>>>(qb, eb, prior);
  // 3) MLP layer 1
  k_mlp1<<<B_ / 16, 256, 0, stream>>>(prior, qf, W1, b1, hbuf);
  // 4) MLP layer 2 + temperature softmax
  k_soft<<<B_ / 16, 256, 0, stream>>>(hbuf, W2, b2, tempp, wts);
  // 5) weighted concept sum + combine + l2norm
  k_dyn<<<B_ / 16, 256, 0, stream>>>(wts, concept, text, alphap, out);
}
